// PAM_Module_37306085933413
// MI455X (gfx1250) — matrix operand order repaired and checked by reading, not compiled, not run on hardware
//
#include <hip/hip_runtime.h>
#include <hip/hip_bf16.h>

// ---------------------------------------------------------------------------
// PAM module on gfx1250: bf16 WMMA (16x16x32) flash-attention, f32 softmax.
// ws layout: Q bf16 [B,N,32] | Kt bf16 [B,N,32] | Vt bf16 [B,C,N]  (18 MiB)
// ---------------------------------------------------------------------------

#define B_    4
#define C_    512
#define N_    4096
#define CQK_  32
#define L2E_  1.44269504088896f

typedef __attribute__((ext_vector_type(16))) __bf16 bf16x16;
typedef __attribute__((ext_vector_type(8)))  __bf16 bf16x8;
typedef __attribute__((ext_vector_type(8)))  float  f32x8;

// ---- WMMA fragment loaders (wave32) ---------------------------------------
// A operand, 16x32 bf16, from row-major [M][K] (ld elems/row):
//   lane L: row = L%16, 16 contiguous K at 16*(L/16)
__device__ __forceinline__ bf16x16 ldfragA_bf16(const __bf16* p, int ld) {
  const int lane = threadIdx.x & 31;
  const int hi = lane >> 4;
  const __bf16* r = p + (size_t)(lane & 15) * ld;
  bf16x8 c0 = *(const bf16x8*)(r + 16 * hi);
  bf16x8 c1 = *(const bf16x8*)(r + 16 * hi + 8);
  bf16x16 a;
#pragma unroll
  for (int i = 0; i < 8; ++i) { a[i] = c0[i]; a[8 + i] = c1[i]; }
  return a;
}

__device__ __forceinline__ bf16x16 ldfragA_f32(const float* p, int ld) {
  const int lane = threadIdx.x & 31;
  const int hi = lane >> 4;
  const float* r = p + (size_t)(lane & 15) * ld;
  f32x8 c0 = *(const f32x8*)(r + 16 * hi);
  f32x8 c1 = *(const f32x8*)(r + 16 * hi + 8);
  bf16x16 a;
#pragma unroll
  for (int i = 0; i < 8; ++i) { a[i] = (__bf16)c0[i]; a[8 + i] = (__bf16)c1[i]; }
  return a;
}

// B operand, 32x16 bf16, from row-major [N][K] (i.e. B^T, K-major):
//   lane L: column = L%16, 16 contiguous K at 16*(L/16)
__device__ __forceinline__ bf16x16 ldfragB_bf16(const __bf16* p, int ld) {
  const int lane = threadIdx.x & 31;
  const __bf16* r = p + (size_t)(lane & 15) * ld + 16 * (lane >> 4);
  return *(const bf16x16*)r;
}

__device__ __forceinline__ bf16x16 ldfragB_f32(const float* p, int ld) {
  const int lane = threadIdx.x & 31;
  const float* r = p + (size_t)(lane & 15) * ld + 16 * (lane >> 4);
  f32x8 c0 = *(const f32x8*)(r);
  f32x8 c1 = *(const f32x8*)(r + 8);
  bf16x16 a;
#pragma unroll
  for (int i = 0; i < 8; ++i) { a[i] = (__bf16)c0[i]; a[8 + i] = (__bf16)c1[i]; }
  return a;
}

__device__ __forceinline__ f32x8 wmma_bf16(bf16x16 a, bf16x16 b, f32x8 c) {
  return __builtin_amdgcn_wmma_f32_16x16x32_bf16(false, a, false, b, (short)0, c,
                                                 false, false);
}

// ---------------------------------------------------------------------------
// Kernel 1: fused projections.  Per block: 64 pixels (n), all of C.
//   Vt[b][c][n] = sum_c' wv[c][c'] x[b][c'][n] + bv[c]     (bf16 out, [C,N])
//   Q [b][n][o] = sum_c' x[b][c'][n] wq[o][c'] + bq[o]     (bf16 out, [N,32])
//   Kt[b][n][o] = sum_c' x[b][c'][n] wk[o][c'] + bk[o]     (bf16 out, [N,32])
// ---------------------------------------------------------------------------
#define XLD 48  // LDS row stride (bf16) for x^T tile: 96B -> 16B & 32B aligned

__global__ void __launch_bounds__(256) pam_proj(
    const float* __restrict__ x,  const float* __restrict__ wq,
    const float* __restrict__ bq, const float* __restrict__ wk,
    const float* __restrict__ bk, const float* __restrict__ wv,
    const float* __restrict__ bv,
    __bf16* __restrict__ Qw, __bf16* __restrict__ Kw, __bf16* __restrict__ Vw) {
  const int b    = blockIdx.y;
  const int n0   = blockIdx.x * 64;
  const int wave = threadIdx.x >> 5;
  const int lane = threadIdx.x & 31;
  const int hi = lane >> 4, lo = lane & 15;

  __shared__ __align__(16) __bf16 Xl[64 * XLD];  // x^T tile: [n][c'] bf16

  f32x8 zero = {};
  f32x8 accV[4][4];
  f32x8 accP[2];
#pragma unroll
  for (int i = 0; i < 4; ++i)
#pragma unroll
    for (int j = 0; j < 4; ++j) accV[i][j] = zero;
  accP[0] = zero; accP[1] = zero;

  const float* xb = x + (size_t)b * C_ * N_;

  for (int kk = 0; kk < C_; kk += 32) {
    // stage x[kk..kk+31][n0..n0+63] transposed into LDS as bf16 [n][c']
#pragma unroll
    for (int i = 0; i < 8; ++i) {
      int e  = threadIdx.x + 256 * i;   // 2048 elements
      int cr = e >> 6, nr = e & 63;
      Xl[nr * XLD + cr] = (__bf16)xb[(size_t)(kk + cr) * N_ + n0 + nr];
    }
    __syncthreads();

    // x^T as B operand for V-proj (4 n-subtiles)
    bf16x16 bx[4];
#pragma unroll
    for (int nt = 0; nt < 4; ++nt) bx[nt] = ldfragB_bf16(Xl + 16 * nt * XLD, XLD);

    // Q (waves 0-3) / K (waves 4-7): A = x^T rows (n-subtile = wave&3)
    bf16x16 ax = ldfragA_bf16(Xl + 16 * (wave & 3) * XLD, XLD);
    const float* wp = (wave < 4) ? wq : wk;
#pragma unroll
    for (int ot = 0; ot < 2; ++ot) {
      bf16x16 bw = ldfragB_f32(wp + (size_t)(16 * ot) * C_ + kk, C_);
      accP[ot] = wmma_bf16(ax, bw, accP[ot]);
    }

    // V-proj: wave owns c rows [64*wave, 64*wave+64)
#pragma unroll
    for (int rt = 0; rt < 4; ++rt) {
      bf16x16 aw = ldfragA_f32(wv + (size_t)(64 * wave + 16 * rt) * C_ + kk, C_);
#pragma unroll
      for (int nt = 0; nt < 4; ++nt)
        accV[rt][nt] = wmma_bf16(aw, bx[nt], accV[rt][nt]);
    }
    __syncthreads();
  }

  // store Vt (bf16, [C][N]) with bias
#pragma unroll
  for (int rt = 0; rt < 4; ++rt)
#pragma unroll
    for (int nt = 0; nt < 4; ++nt)
#pragma unroll
      for (int r = 0; r < 8; ++r) {
        int c = 64 * wave + 16 * rt + r + 8 * hi;
        int n = n0 + 16 * nt + lo;
        Vw[(size_t)b * C_ * N_ + (size_t)c * N_ + n] =
            (__bf16)(accV[rt][nt][r] + bv[c]);
      }

  // store Q / Kt (bf16, [N][32]) with bias
  __bf16* qk       = (wave < 4) ? Qw : Kw;
  const float* bia = (wave < 4) ? bq : bk;
#pragma unroll
  for (int ot = 0; ot < 2; ++ot)
#pragma unroll
    for (int r = 0; r < 8; ++r) {
      int n = n0 + 16 * (wave & 3) + r + 8 * hi;
      int o = 16 * ot + lo;
      qk[(size_t)b * N_ * CQK_ + (size_t)n * CQK_ + o] = (__bf16)(accP[ot][r] + bia[o]);
    }
}

// ---------------------------------------------------------------------------
// Kernel 2: flash attention + residual.
// Block: 64 i-rows of one batch, 8 waves.
//   waves 0-3: S = Q·K^T (bf16 WMMA), online softmax -> P(bf16), alpha to LDS
//   all waves: O[64 x 64c-chunk] = alpha*O + P·V  (bf16 WMMA, f32 accum)
// Epilogue: out = gamma * O/l + x  (f32).
// ---------------------------------------------------------------------------
#define PLD 80  // LDS row stride (bf16) for P tile: 160B -> 16B aligned chunks

__global__ void __launch_bounds__(256) pam_attn(
    const float* __restrict__ x, const float* __restrict__ gamma,
    const __bf16* __restrict__ Qw, const __bf16* __restrict__ Kw,
    const __bf16* __restrict__ Vw, float* __restrict__ out) {
  const int b    = blockIdx.y;
  const int i0   = blockIdx.x * 64;
  const int wave = threadIdx.x >> 5;
  const int lane = threadIdx.x & 31;
  const int hi = lane >> 4, lo = lane & 15;

  __shared__ __align__(16) __bf16 Pl[64 * PLD];
  __shared__ float alpha_s[64];
  __shared__ float l_s[64];

  const __bf16* Qb = Qw + (size_t)b * N_ * CQK_;
  const __bf16* Kb = Kw + (size_t)b * N_ * CQK_;
  const __bf16* Vb = Vw + (size_t)b * C_ * N_;

  f32x8 zero = {};
  f32x8 accO[4][4];
#pragma unroll
  for (int i = 0; i < 4; ++i)
#pragma unroll
    for (int j = 0; j < 4; ++j) accO[i][j] = zero;

  bf16x16 aQ = {};
  float m_run[8], l_run[8];
#pragma unroll
  for (int r = 0; r < 8; ++r) { m_run[r] = -3.0e38f; l_run[r] = 0.f; }
  if (wave < 4) aQ = ldfragA_bf16(Qb + (size_t)(i0 + 16 * wave) * CQK_, CQK_);

  for (int j0 = 0; j0 < N_; j0 += 64) {
    if (j0 + 64 < N_) {
      __builtin_prefetch(Kb + (size_t)(j0 + 64) * CQK_ + lane * 2, 0, 0);
    }
    if (wave < 4) {
      // ---- S = Q · K^T  (one 16x16x32 WMMA per n-subtile, K depth = 32)
      f32x8 S[4];
#pragma unroll
      for (int nt = 0; nt < 4; ++nt) {
        bf16x16 bK = ldfragB_bf16(Kb + (size_t)(j0 + 16 * nt) * CQK_, CQK_);
        S[nt] = wmma_bf16(aQ, bK, zero);
      }
      // ---- online softmax over 64 columns; row = 16*wave + r + 8*hi
      float mnew[8], al[8], rsum[8];
#pragma unroll
      for (int r = 0; r < 8; ++r) {
        float v = fmaxf(fmaxf(S[0][r], S[1][r]), fmaxf(S[2][r], S[3][r]));
#pragma unroll
        for (int off = 1; off < 16; off <<= 1) v = fmaxf(v, __shfl_xor(v, off, 16));
        mnew[r] = fmaxf(m_run[r], v);
        al[r]   = exp2f((m_run[r] - mnew[r]) * L2E_);
        rsum[r] = 0.f;
      }
#pragma unroll
      for (int nt = 0; nt < 4; ++nt)
#pragma unroll
        for (int r = 0; r < 8; ++r) {
          float p = exp2f((S[nt][r] - mnew[r]) * L2E_);
          S[nt][r] = p;
          rsum[r] += p;
        }
#pragma unroll
      for (int r = 0; r < 8; ++r) {
        float v = rsum[r];
#pragma unroll
        for (int off = 1; off < 16; off <<= 1) v += __shfl_xor(v, off, 16);
        l_run[r] = l_run[r] * al[r] + v;
        m_run[r] = mnew[r];
      }
      // ---- publish P (bf16) and alpha
#pragma unroll
      for (int nt = 0; nt < 4; ++nt)
#pragma unroll
        for (int r = 0; r < 8; ++r)
          Pl[(16 * wave + r + 8 * hi) * PLD + 16 * nt + lo] = (__bf16)S[nt][r];
      if (lo == 0)
#pragma unroll
        for (int r = 0; r < 8; ++r) alpha_s[16 * wave + r + 8 * hi] = al[r];
    }
    __syncthreads();

    // ---- O = alpha*O + P · V ; wave owns c-chunk [64*wave, 64*wave+64)
#pragma unroll
    for (int rt = 0; rt < 4; ++rt) {
      bf16x16 A0 = ldfragA_bf16(Pl + (size_t)(16 * rt) * PLD, PLD);
      bf16x16 A1 = ldfragA_bf16(Pl + (size_t)(16 * rt) * PLD + 32, PLD);
#pragma unroll
      for (int r = 0; r < 8; ++r) {
        float a = alpha_s[16 * rt + r + 8 * hi];
#pragma unroll
        for (int ct = 0; ct < 4; ++ct) accO[rt][ct][r] *= a;
      }
#pragma unroll
      for (int ct = 0; ct < 4; ++ct) {
        const __bf16* vrow = Vb + (size_t)(64 * wave + 16 * ct) * N_ + j0;
        bf16x16 bV0 = ldfragB_bf16(vrow, N_);
        bf16x16 bV1 = ldfragB_bf16(vrow + 32, N_);
        accO[rt][ct] = wmma_bf16(A0, bV0, accO[rt][ct]);
        accO[rt][ct] = wmma_bf16(A1, bV1, accO[rt][ct]);
      }
    }
    __syncthreads();
  }

  if (wave < 4 && lo == 0)
#pragma unroll
    for (int r = 0; r < 8; ++r) l_s[16 * wave + r + 8 * hi] = l_run[r];
  __syncthreads();

  // ---- epilogue: out[b][c][i] = gamma * O[i][c]/l[i] + x[b][c][i]
  const float g   = gamma[0];
  const float* xb = x + (size_t)b * C_ * N_;
  float* ob       = out + (size_t)b * C_ * N_;
#pragma unroll
  for (int rt = 0; rt < 4; ++rt)
#pragma unroll
    for (int r = 0; r < 8; ++r) {
      int i = i0 + 16 * rt + r + 8 * hi;
      float linv = 1.f / l_s[16 * rt + r + 8 * hi];
#pragma unroll
      for (int ct = 0; ct < 4; ++ct) {
        int c = 64 * wave + 16 * ct + lo;
        size_t idx = (size_t)c * N_ + i;
        ob[idx] = g * accO[rt][ct][r] * linv + xb[idx];
      }
    }
}

// ---------------------------------------------------------------------------
extern "C" void kernel_launch(void* const* d_in, const int* in_sizes, int n_in,
                              void* d_out, int out_size, void* d_ws, size_t ws_size,
                              hipStream_t stream) {
  (void)in_sizes; (void)n_in; (void)out_size; (void)ws_size;
  const float* x     = (const float*)d_in[0];
  const float* wq    = (const float*)d_in[1];
  const float* bq    = (const float*)d_in[2];
  const float* wk    = (const float*)d_in[3];
  const float* bk    = (const float*)d_in[4];
  const float* wv    = (const float*)d_in[5];
  const float* bv    = (const float*)d_in[6];
  const float* gamma = (const float*)d_in[7];
  float* out = (float*)d_out;

  __bf16* Qw = (__bf16*)d_ws;                  // B*N*32 bf16 = 1 MiB
  __bf16* Kw = Qw + (size_t)B_ * N_ * CQK_;    // 1 MiB
  __bf16* Vw = Kw + (size_t)B_ * N_ * CQK_;    // B*C*N bf16 = 16 MiB

  dim3 grid(N_ / 64, B_);
  pam_proj<<<grid, 256, 0, stream>>>(x, wq, bq, wk, bk, wv, bv, Qw, Kw, Vw);
  pam_attn<<<grid, 256, 0, stream>>>(x, gamma, Qw, Kw, Vw, out);
}
